// UniMolLayer_38732015075514
// MI455X (gfx1250) — matrix operand order repaired and checked by reading, not compiled, not run on hardware
//
#include <hip/hip_runtime.h>
#include <math.h>

// ---------------------------------------------------------------------------
// UniMol layer for MI455X (gfx1250, wave32, WMMA, async global->LDS copies).
// bf16 matrix-core GEMMs with f32 accumulation; softmax/LN in f32.
// ---------------------------------------------------------------------------

typedef __bf16 bf16;
typedef __bf16 v16bf  __attribute__((ext_vector_type(16)));
typedef __bf16 bf16x8 __attribute__((ext_vector_type(8)));
typedef float  v8f    __attribute__((ext_vector_type(8)));
typedef int    v4i    __attribute__((ext_vector_type(4)));

#define B_   2
#define L_   1024
#define D_   1024
#define H_   16
#define HD_  64
#define FFN_ 2048
#define M_   (B_ * L_)   // 2048 token rows

enum { EPI_F32 = 0, EPI_BF16 = 1, EPI_SILU = 2 };

// --------------------- CDNA5 async global->LDS helpers ---------------------
// 16B per-lane DMA into LDS, tracked by ASYNCcnt (no VGPR round trip).
// Builtin signature (from compiler feedback): (AS1 v4i*, AS3 v4i*, Imm, Imm).

typedef __attribute__((address_space(1))) v4i* gv4i_p;
typedef __attribute__((address_space(3))) v4i* lv4i_p;

__device__ __forceinline__ void async_cp16(void* lds, const void* gptr) {
#if defined(__gfx1250__) && __has_builtin(__builtin_amdgcn_global_load_async_to_lds_b128)
  __builtin_amdgcn_global_load_async_to_lds_b128(
      (gv4i_p)(void*)gptr, (lv4i_p)lds, 0, 0);
#else
  *(bf16x8*)lds = *(const bf16x8*)gptr;
#endif
}

__device__ __forceinline__ void wait_async_all() {
#if defined(__gfx1250__) && __has_builtin(__builtin_amdgcn_s_wait_asynccnt)
  __builtin_amdgcn_s_wait_asynccnt(0);
#elif defined(__gfx1250__)
  asm volatile("s_wait_asynccnt 0x0" ::: "memory");
#endif
}

// ------------------------------ elementwise --------------------------------

__global__ void cvt_f2b(const float* __restrict__ in, bf16* __restrict__ out, long n) {
  long i = ((long)blockIdx.x * blockDim.x + threadIdx.x) * 4;
  if (i + 3 < n) {
    float4 v = *(const float4*)(in + i);
    out[i + 0] = (bf16)v.x; out[i + 1] = (bf16)v.y;
    out[i + 2] = (bf16)v.z; out[i + 3] = (bf16)v.w;
  }
}

__global__ void gate_mul(const float* __restrict__ g, const float* __restrict__ o,
                         bf16* __restrict__ out, long n) {
  long i = ((long)blockIdx.x * blockDim.x + threadIdx.x) * 4;
  if (i + 3 < n) {
#pragma unroll
    for (int r = 0; r < 4; ++r) {
      float gv = g[i + r];
      float s  = 1.f / (1.f + __expf(-gv));
      out[i + r] = (bf16)(s * o[i + r]);
    }
  }
}

// ----------------------------- transposes ----------------------------------

// out[c][r] = (bf16) in[r][c] ; R,C multiples of 32
__global__ void transpose_f2b(const float* __restrict__ in, bf16* __restrict__ out,
                              int R, int C) {
  __shared__ float tile[32][33];
  const int c0 = blockIdx.x * 32, r0 = blockIdx.y * 32;
  const int tx = threadIdx.x, ty = threadIdx.y;
#pragma unroll
  for (int i = 0; i < 4; ++i)
    tile[ty + i * 8][tx] = in[(long)(r0 + ty + i * 8) * C + c0 + tx];
  __syncthreads();
#pragma unroll
  for (int i = 0; i < 4; ++i)
    out[(long)(c0 + ty + i * 8) * R + r0 + tx] = (bf16)tile[tx][ty + i * 8];
}

// batched bf16 transpose: per z = b*HB+h, in += b*sib + h*sih ; out += z*sob
__global__ void transpose_b2b(const bf16* __restrict__ in, long ld_in, long sib, long sih,
                              int HB, bf16* __restrict__ out, long ld_out, long sob) {
  __shared__ bf16 tile[32][33];
  const int z = blockIdx.z, b = z / HB, h = z % HB;
  in  += (long)b * sib + (long)h * sih;
  out += (long)z * sob;
  const int c0 = blockIdx.x * 32, r0 = blockIdx.y * 32;
  const int tx = threadIdx.x, ty = threadIdx.y;
#pragma unroll
  for (int i = 0; i < 4; ++i)
    tile[ty + i * 8][tx] = in[(long)(r0 + ty + i * 8) * ld_in + c0 + tx];
  __syncthreads();
#pragma unroll
  for (int i = 0; i < 4; ++i)
    out[(long)(c0 + ty + i * 8) * ld_out + r0 + tx] = tile[tx][ty + i * 8];
}

// ------------------------------ WMMA GEMM ----------------------------------
// C[m,n] = sum_k A[m,k] * Bt[n,k]   (bf16 inputs, f32 accum)
// Block tile 128x64x32, 8 waves (wave32), each wave: 2x2 x (16x16) WMMA tiles.
// Double-buffered LDS; tiles staged with async global->LDS B128 copies so the
// next tile's DMA overlaps ds_load + v_wmma of the current tile.
// Batched via z = b*HB + h with per-operand (b,h) strides.

template <int EPI>
__global__ __launch_bounds__(256) void wmma_gemm(
    const bf16* __restrict__ A, long lda, long sAb, long sAh,
    const bf16* __restrict__ Bt, long ldb, long sBb, long sBh,
    void* __restrict__ Cv, long ldcm, long ldcn, long sCb, long sCh,
    const float* __restrict__ bias, float scale, int K, int HB) {
  __shared__ bf16 As[2][128][40];  // +8 pad keeps 16B alignment, spreads banks
  __shared__ bf16 Bs[2][64][40];

  const int t = threadIdx.x;
  const int lane = t & 31, wave = t >> 5;
  const int waveM = (wave >> 1) * 32, waveN = (wave & 1) * 32;
  const long m0 = (long)blockIdx.y * 128, n0 = (long)blockIdx.x * 64;
  const int z = blockIdx.z, bz = z / HB, hz = z % HB;
  A  += (long)bz * sAb + (long)hz * sAh;
  Bt += (long)bz * sBb + (long)hz * sBh;
  const long cbase = (long)bz * sCb + (long)hz * sCh;

  v8f acc[2][2] = {};

  const int arow = t >> 1, acol = (t & 1) * 16;  // A: 128 rows x 32 k
  const int brow = t >> 2, bcol = (t & 3) * 8;   // B: 64 rows x 32 k
  const int laneN = lane & 15;
  const int kbA = (lane >> 4) * 8;   // A frag: lanes>=16 hold K+8 halves
  const int kbB = (lane >> 4) * 8;   // B frag: lanes>=16 hold K+8 halves
  const bf16* aSrc = A + (m0 + arow) * lda + acol;
  const bf16* bSrc = Bt + (n0 + brow) * ldb + bcol;

  // prologue: stage k0 = 0 into buffer 0
  async_cp16(&As[0][arow][acol],     aSrc);
  async_cp16(&As[0][arow][acol + 8], aSrc + 8);
  async_cp16(&Bs[0][brow][bcol],     bSrc);
  wait_async_all();
  __syncthreads();

  int cur = 0;
  for (int k0 = 0; k0 < K; k0 += 32) {
    // kick off next tile's DMA into the other buffer
    if (k0 + 32 < K) {
      const int nxt = cur ^ 1;
      async_cp16(&As[nxt][arow][acol],     aSrc + k0 + 32);
      async_cp16(&As[nxt][arow][acol + 8], aSrc + k0 + 40);
      async_cp16(&Bs[nxt][brow][bcol],     bSrc + k0 + 32);
    }

    // fragments per ISA 7.12.2 lane layouts, then 4 WMMAs
    union Frag { v16bf v; bf16x8 h[2]; } afr[2], bfr[2];
#pragma unroll
    for (int mi = 0; mi < 2; ++mi) {
      int r = waveM + mi * 16 + laneN;
      afr[mi].h[0] = *(const bf16x8*)&As[cur][r][kbA];
      afr[mi].h[1] = *(const bf16x8*)&As[cur][r][kbA + 16];
    }
#pragma unroll
    for (int ni = 0; ni < 2; ++ni) {
      int c = waveN + ni * 16 + laneN;
      bfr[ni].h[0] = *(const bf16x8*)&Bs[cur][c][kbB];
      bfr[ni].h[1] = *(const bf16x8*)&Bs[cur][c][kbB + 16];
    }
#pragma unroll
    for (int mi = 0; mi < 2; ++mi)
#pragma unroll
      for (int ni = 0; ni < 2; ++ni)
        acc[mi][ni] = __builtin_amdgcn_wmma_f32_16x16x32_bf16(
            false, afr[mi].v, false, bfr[ni].v, (short)0, acc[mi][ni], false, false);

    wait_async_all();   // next tile landed in LDS
    __syncthreads();    // everyone done reading cur; staged data visible
    cur ^= 1;
  }

  // ---- epilogue (C/D layout: VGPR j -> M = j + 8*(lane>=16), N = lane&15) ----
  const int laneHalf = lane >> 4;
#pragma unroll
  for (int mi = 0; mi < 2; ++mi) {
#pragma unroll
    for (int ni = 0; ni < 2; ++ni) {
      const long nn = n0 + waveN + ni * 16 + laneN;
      const float bv = bias ? bias[nn] : 0.f;
#pragma unroll
      for (int j = 0; j < 8; ++j) {
        const long mm = m0 + waveM + mi * 16 + j + laneHalf * 8;
        const long idx = cbase + mm * ldcm + nn * ldcn;
        const float v = acc[mi][ni][j];
        if (EPI == EPI_F32) {
          ((float*)Cv)[idx] = v * scale + bv;
        } else if (EPI == EPI_BF16) {
          ((bf16*)Cv)[idx] = (bf16)(v * scale + bv);
        } else {  // EPI_SILU
          float u = v + bv;
          ((bf16*)Cv)[idx] = (bf16)(u / (1.f + __expf(-u)));
        }
      }
    }
  }
}

// ----------------------------- reductions ----------------------------------

__device__ __forceinline__ float blk_sum(float v, float* sm) {
  int t = threadIdx.x;
  sm[t] = v; __syncthreads();
  for (int s = 128; s > 0; s >>= 1) { if (t < s) sm[t] += sm[t + s]; __syncthreads(); }
  float r = sm[0]; __syncthreads();
  return r;
}
__device__ __forceinline__ float blk_max(float v, float* sm) {
  int t = threadIdx.x;
  sm[t] = v; __syncthreads();
  for (int s = 128; s > 0; s >>= 1) { if (t < s) sm[t] = fmaxf(sm[t], sm[t + s]); __syncthreads(); }
  float r = sm[0]; __syncthreads();
  return r;
}

// Softmax over keys for one (b,h,q) row.
//   S:    f32 scores [b,h,q,k]  (contiguous row read)
//   eb:   f32 bias   [b,q,k,h]  (stride-H read; h-sibling blocks adjacent -> L2 merge)
//   attn: f32 output [b,q,k,h]  (stride-H write; merged in L2 across h-siblings)
//   Pb:   bf16 probs [b,h,q,k]  (contiguous write, feeds PV GEMM)
// grid = (H, L, B) so the 16 h-blocks of one (b,q) are consecutive.
__global__ __launch_bounds__(256) void softmax_keys(
    const float* __restrict__ S, const float* __restrict__ eb,
    float* __restrict__ attn, bf16* __restrict__ Pb,
    const unsigned char* __restrict__ mask) {
  __shared__ float sm[256];
  const int h = blockIdx.x, q = blockIdx.y, b = blockIdx.z;
  const float* srow = S + ((long)(b * H_ + h) * L_ + q) * L_;
  const float* erow = eb + ((long)(b * L_ + q) * L_) * H_ + h;
  float* arow = attn + ((long)(b * L_ + q) * L_) * H_ + h;
  bf16* prow = Pb + ((long)(b * H_ + h) * L_ + q) * L_;

  float vals[4];
  float mx = -__builtin_inff();
#pragma unroll
  for (int r = 0; r < 4; ++r) {
    int k = threadIdx.x + r * 256;
    float v = srow[k] + erow[(long)k * H_];
    if (mask[b * L_ + k]) v = -__builtin_inff();
    vals[r] = v;
    mx = fmaxf(mx, v);
  }
  mx = blk_max(mx, sm);
  float s = 0.f;
#pragma unroll
  for (int r = 0; r < 4; ++r) { float p = __expf(vals[r] - mx); vals[r] = p; s += p; }
  s = blk_sum(s, sm);
  const float inv = 1.f / s;
#pragma unroll
  for (int r = 0; r < 4; ++r) {
    int k = threadIdx.x + r * 256;
    float p = vals[r] * inv;
    arow[(long)k * H_] = p;
    prow[k] = (bf16)p;
  }
}

// y = LayerNorm(x + dx) * g + b ; writes f32 and optional bf16 copy
__global__ __launch_bounds__(256) void add_layernorm(
    const float* __restrict__ x, const float* __restrict__ dx,
    const float* __restrict__ g, const float* __restrict__ bta,
    float* __restrict__ outf, bf16* __restrict__ outb) {
  __shared__ float sm[256];
  const long row = blockIdx.x;
  const float* xr = x + row * D_;
  const float* dr = dx + row * D_;
  float v[4], s = 0.f;
#pragma unroll
  for (int r = 0; r < 4; ++r) {
    int i = threadIdx.x + r * 256;
    v[r] = xr[i] + dr[i];
    s += v[r];
  }
  s = blk_sum(s, sm);
  const float mean = s * (1.f / D_);
  float sq = 0.f;
#pragma unroll
  for (int r = 0; r < 4; ++r) { float d = v[r] - mean; sq += d * d; }
  sq = blk_sum(sq, sm);
  const float rstd = rsqrtf(sq * (1.f / D_) + 1e-5f);
#pragma unroll
  for (int r = 0; r < 4; ++r) {
    int i = threadIdx.x + r * 256;
    float y = (v[r] - mean) * rstd * g[i] + bta[i];
    outf[row * D_ + i] = y;
    if (outb) outb[row * D_ + i] = (bf16)y;
  }
}

// ------------------------------- driver ------------------------------------

extern "C" void kernel_launch(void* const* d_in, const int* in_sizes, int n_in,
                              void* d_out, int out_size, void* d_ws, size_t ws_size,
                              hipStream_t stream) {
  (void)in_sizes; (void)n_in; (void)out_size; (void)ws_size;
  const float* x    = (const float*)d_in[0];
  const float* eb   = (const float*)d_in[1];
  const unsigned char* mask = (const unsigned char*)d_in[2];
  const float* wq   = (const float*)d_in[3];
  const float* wk   = (const float*)d_in[4];
  const float* wv   = (const float*)d_in[5];
  const float* wo   = (const float*)d_in[6];
  const float* bo   = (const float*)d_in[7];
  const float* wg   = (const float*)d_in[8];
  const float* bg   = (const float*)d_in[9];
  const float* w1   = (const float*)d_in[10];
  const float* b1   = (const float*)d_in[11];
  const float* w2   = (const float*)d_in[12];
  const float* b2   = (const float*)d_in[13];
  const float* ln1g = (const float*)d_in[14];
  const float* ln1b = (const float*)d_in[15];
  const float* ln2g = (const float*)d_in[16];
  const float* ln2b = (const float*)d_in[17];

  char* ws = (char*)d_ws;
  size_t off = 0;
  auto alloc = [&](size_t bytes) -> void* {
    void* p = ws + off;
    off += (bytes + 255) & ~(size_t)255;
    return p;
  };

  bf16* xb  = (bf16*)alloc((size_t)M_ * D_ * 2);
  bf16* wqt = (bf16*)alloc((size_t)D_ * D_ * 2);
  bf16* wkt = (bf16*)alloc((size_t)D_ * D_ * 2);
  bf16* wvt = (bf16*)alloc((size_t)D_ * D_ * 2);
  bf16* wgt = (bf16*)alloc((size_t)D_ * D_ * 2);
  bf16* wot = (bf16*)alloc((size_t)D_ * D_ * 2);
  bf16* w1t = (bf16*)alloc((size_t)D_ * FFN_ * 2);
  bf16* w2t = (bf16*)alloc((size_t)FFN_ * D_ * 2);
  bf16* qB  = (bf16*)alloc((size_t)M_ * D_ * 2);
  bf16* kB  = (bf16*)alloc((size_t)M_ * D_ * 2);
  bf16* vB  = (bf16*)alloc((size_t)M_ * D_ * 2);
  bf16* vT  = (bf16*)alloc((size_t)B_ * H_ * HD_ * L_ * 2);
  bf16* Pb  = (bf16*)alloc((size_t)B_ * H_ * L_ * L_ * 2);
  float* Sf = (float*)alloc((size_t)B_ * H_ * L_ * L_ * 4);  // scores [b,h,q,k]
  float* gf    = (float*)alloc((size_t)M_ * D_ * 4);
  float* of    = (float*)alloc((size_t)M_ * D_ * 4);
  bf16*  ob    = (bf16*) alloc((size_t)M_ * D_ * 2);
  float* delta = (float*)alloc((size_t)M_ * D_ * 4);
  float* yf    = (float*)alloc((size_t)M_ * D_ * 4);
  bf16*  yb    = (bf16*) alloc((size_t)M_ * D_ * 2);
  bf16*  hb    = (bf16*) alloc((size_t)M_ * FFN_ * 2);
  float* ffn   = (float*)alloc((size_t)M_ * D_ * 4);

  float* attn = (float*)d_out + (size_t)M_ * D_;  // [b,q,k,h] region of d_out

  const dim3 tb(32, 8);

  // bf16 conversions / weight transposes
  cvt_f2b<<<(M_ * D_) / 1024, 256, 0, stream>>>(x, xb, (long)M_ * D_);
  transpose_f2b<<<dim3(D_ / 32, D_ / 32), tb, 0, stream>>>(wq, wqt, D_, D_);
  transpose_f2b<<<dim3(D_ / 32, D_ / 32), tb, 0, stream>>>(wk, wkt, D_, D_);
  transpose_f2b<<<dim3(D_ / 32, D_ / 32), tb, 0, stream>>>(wv, wvt, D_, D_);
  transpose_f2b<<<dim3(D_ / 32, D_ / 32), tb, 0, stream>>>(wg, wgt, D_, D_);
  transpose_f2b<<<dim3(D_ / 32, D_ / 32), tb, 0, stream>>>(wo, wot, D_, D_);
  transpose_f2b<<<dim3(FFN_ / 32, D_ / 32), tb, 0, stream>>>(w1, w1t, D_, FFN_);
  transpose_f2b<<<dim3(D_ / 32, FFN_ / 32), tb, 0, stream>>>(w2, w2t, FFN_, D_);

  // Q (scaled 1/sqrt(64)), K, V, gate projections
  const dim3 gP(D_ / 64, M_ / 128, 1);
  wmma_gemm<EPI_BF16><<<gP, 256, 0, stream>>>(xb, D_, 0, 0, wqt, D_, 0, 0,
      qB, D_, 1, 0, 0, nullptr, 0.125f, D_, 1);
  wmma_gemm<EPI_BF16><<<gP, 256, 0, stream>>>(xb, D_, 0, 0, wkt, D_, 0, 0,
      kB, D_, 1, 0, 0, nullptr, 1.f, D_, 1);
  wmma_gemm<EPI_BF16><<<gP, 256, 0, stream>>>(xb, D_, 0, 0, wvt, D_, 0, 0,
      vB, D_, 1, 0, 0, nullptr, 1.f, D_, 1);
  wmma_gemm<EPI_F32><<<gP, 256, 0, stream>>>(xb, D_, 0, 0, wgt, D_, 0, 0,
      gf, D_, 1, 0, 0, bg, 1.f, D_, 1);

  // scores = Q K^T -> Sf [b,h,q,k] (fully coalesced epilogue)
  const dim3 gS(L_ / 64, L_ / 128, B_ * H_);
  wmma_gemm<EPI_F32><<<gS, 256, 0, stream>>>(
      qB, D_, (long)L_ * D_, HD_,
      kB, D_, (long)L_ * D_, HD_,
      Sf, L_, 1, (long)H_ * L_ * L_, (long)L_ * L_,
      nullptr, 1.f, HD_, H_);

  // softmax over keys (+edge_bias, +mask); writes attn_out and bf16 probs
  softmax_keys<<<dim3(H_, L_, B_), 256, 0, stream>>>(Sf, eb, attn, Pb, mask);

  // V^T per (b,h): [HD x L]
  transpose_b2b<<<dim3(HD_ / 32, L_ / 32, B_ * H_), tb, 0, stream>>>(
      vB, D_, (long)L_ * D_, HD_, H_, vT, L_, (long)HD_ * L_);

  // O = P V  -> [b,l,h,d] f32
  const dim3 gPV(HD_ / 64, L_ / 128, B_ * H_);
  wmma_gemm<EPI_F32><<<gPV, 256, 0, stream>>>(
      Pb, L_, (long)H_ * L_ * L_, (long)L_ * L_,
      vT, L_, (long)H_ * HD_ * L_, (long)HD_ * L_,
      of, D_, 1, (long)L_ * D_, HD_,
      nullptr, 1.f, L_, H_);

  // sigmoid gate, output projection, residual + LN1
  gate_mul<<<(M_ * D_) / 1024, 256, 0, stream>>>(gf, of, ob, (long)M_ * D_);
  wmma_gemm<EPI_F32><<<gP, 256, 0, stream>>>(ob, D_, 0, 0, wot, D_, 0, 0,
      delta, D_, 1, 0, 0, bo, 1.f, D_, 1);
  add_layernorm<<<M_, 256, 0, stream>>>(x, delta, ln1g, ln1b, yf, yb);

  // FFN: silu(y w1 + b1) w2 + b2, residual + LN2 -> d_out x-region
  const dim3 gF1(FFN_ / 64, M_ / 128, 1);
  wmma_gemm<EPI_SILU><<<gF1, 256, 0, stream>>>(yb, D_, 0, 0, w1t, D_, 0, 0,
      hb, FFN_, 1, 0, 0, b1, 1.f, D_, 1);
  wmma_gemm<EPI_F32><<<gP, 256, 0, stream>>>(hb, FFN_, 0, 0, w2t, FFN_, 0, 0,
      ffn, D_, 1, 0, 0, b2, 1.f, FFN_, 1);
  add_layernorm<<<M_, 256, 0, stream>>>(yf, ffn, ln2g, ln2b, (float*)d_out, nullptr);
}
